// PointViT_52003464020192
// MI455X (gfx1250) — matrix operand order repaired and checked by reading, not compiled, not run on hardware
//
#include <hip/hip_runtime.h>
#include <hip/hip_bf16.h>
#include <hip/hip_fp16.h>

#define BB    32
#define NN    2048
#define NSAMP 512
#define KNBR  16
#define MTOT  (BB * NSAMP * KNBR)   // 262144 positions
#define GTOT  (BB * NSAMP)          // 16384 groups
#define EPSV  1e-5f

typedef _Float16 h8  __attribute__((ext_vector_type(8)));
typedef _Float16 h16 __attribute__((ext_vector_type(16)));
typedef float    f8  __attribute__((ext_vector_type(8)));

// ---------------------------------------------------------------------------
// Small utility kernels
// ---------------------------------------------------------------------------
__global__ void zero_kernel(float* buf, int n) {
    int i = blockIdx.x * 256 + threadIdx.x;
    if (i < n) buf[i] = 0.0f;
}

__global__ void cvt_kernel(const float* __restrict__ src, _Float16* __restrict__ dst, int n) {
    int i = blockIdx.x * 256 + threadIdx.x;
    if (i < n) dst[i] = (_Float16)src[i];
}

// Wce[o][c] = sum_i w2[o][i] * (w1[i][c] + w1[i][c+3]),  o in [0,256), c in [0,3)
__global__ void wce_kernel(const float* __restrict__ w1, const float* __restrict__ w2,
                           float* __restrict__ wce) {
    int o = threadIdx.x;
    float a0 = 0.f, a1 = 0.f, a2 = 0.f;
    for (int i = 0; i < 256; i++) {
        float wv = w2[o * 256 + i];
        a0 += wv * (w1[i * 6 + 0] + w1[i * 6 + 3]);
        a1 += wv * (w1[i * 6 + 1] + w1[i * 6 + 4]);
        a2 += wv * (w1[i * 6 + 2] + w1[i * 6 + 5]);
    }
    wce[o * 3 + 0] = a0;
    wce[o * 3 + 1] = a1;
    wce[o * 3 + 2] = a2;
}

// ---------------------------------------------------------------------------
// Farthest point sampling: one block per batch element. Writes cntrd (B,NS,3).
// ---------------------------------------------------------------------------
__global__ __launch_bounds__(256) void fps_kernel(const float* __restrict__ p,
                                                  float* __restrict__ cntrd) {
    __shared__ float px[NN], py[NN], pz[NN], dist[NN];
    __shared__ float rv[256];
    __shared__ int   ri[256];
    __shared__ int   sfar;
    const int b = blockIdx.x, t = threadIdx.x;
    const float* pb = p + (size_t)b * NN * 3;
    for (int i = t; i < NN; i += 256) {
        px[i] = pb[i * 3 + 0];
        py[i] = pb[i * 3 + 1];
        pz[i] = pb[i * 3 + 2];
        dist[i] = 1e10f;
    }
    if (t == 0) sfar = 0;
    __syncthreads();
    for (int s = 0; s < NSAMP; s++) {
        const int far = sfar;
        const float cx = px[far], cy = py[far], cz = pz[far];
        if (t == 0) {
            float* o = cntrd + ((size_t)b * NSAMP + s) * 3;
            o[0] = cx; o[1] = cy; o[2] = cz;
        }
        float best = -1.0f; int bidx = 0;
        for (int i = t; i < NN; i += 256) {
            float dx = px[i] - cx, dy = py[i] - cy, dz = pz[i] - cz;
            float d = dx * dx + dy * dy + dz * dz;
            float m = fminf(dist[i], d);
            dist[i] = m;
            if (m > best) { best = m; bidx = i; }
        }
        rv[t] = best; ri[t] = bidx;
        __syncthreads();
        for (int st = 128; st > 0; st >>= 1) {
            if (t < st) {
                if (rv[t + st] > rv[t] || (rv[t + st] == rv[t] && ri[t + st] < ri[t])) {
                    rv[t] = rv[t + st]; ri[t] = ri[t + st];
                }
            }
            __syncthreads();
        }
        if (t == 0) sfar = ri[0];
        __syncthreads();
    }
}

// ---------------------------------------------------------------------------
// KNN: block handles one batch x 16 centroids; 16-way iterative argmin.
// ---------------------------------------------------------------------------
__global__ __launch_bounds__(256) void knn_kernel(const float* __restrict__ p,
                                                  const float* __restrict__ cntrd,
                                                  int* __restrict__ nidx) {
    __shared__ float px[NN], py[NN], pz[NN], dist[NN];
    __shared__ float rv[256];
    __shared__ int   ri[256];
    const int b = blockIdx.x >> 5;
    const int chunk = blockIdx.x & 31;
    const int t = threadIdx.x;
    const float* pb = p + (size_t)b * NN * 3;
    for (int i = t; i < NN; i += 256) {
        px[i] = pb[i * 3 + 0];
        py[i] = pb[i * 3 + 1];
        pz[i] = pb[i * 3 + 2];
    }
    __syncthreads();
    for (int sc = 0; sc < 16; sc++) {
        const int s = chunk * 16 + sc;
        const float* c = cntrd + ((size_t)b * NSAMP + s) * 3;
        const float cx = c[0], cy = c[1], cz = c[2];
        for (int i = t; i < NN; i += 256) {
            float dx = px[i] - cx, dy = py[i] - cy, dz = pz[i] - cz;
            dist[i] = dx * dx + dy * dy + dz * dz;
        }
        __syncthreads();
        for (int kk = 0; kk < KNBR; kk++) {
            float best = 1e30f; int bidx = 0;
            for (int i = t; i < NN; i += 256) {
                float d = dist[i];
                if (d < best) { best = d; bidx = i; }
            }
            rv[t] = best; ri[t] = bidx;
            __syncthreads();
            for (int st = 128; st > 0; st >>= 1) {
                if (t < st) {
                    if (rv[t + st] < rv[t] || (rv[t + st] == rv[t] && ri[t + st] < ri[t])) {
                        rv[t] = rv[t + st]; ri[t] = ri[t + st];
                    }
                }
                __syncthreads();
            }
            if (t == 0) {
                nidx[((size_t)b * NSAMP + s) * KNBR + kk] = ri[0];
                dist[ri[0]] = 1e30f;
            }
            __syncthreads();
        }
    }
}

// ---------------------------------------------------------------------------
// Stage 1 stats: h2 = Wce @ g + b2 recomputed on the fly; per-channel sum/sumsq.
// Block handles 256 positions; thread t == channel t.
// ---------------------------------------------------------------------------
__global__ __launch_bounds__(256) void stats1_kernel(const float* __restrict__ p,
                                                     const int* __restrict__ nidx,
                                                     const float* __restrict__ wce,
                                                     const float* __restrict__ bias2,
                                                     float* __restrict__ sum1,
                                                     float* __restrict__ ss1) {
    __shared__ float gx[256], gy[256], gz[256];
    const int t = threadIdx.x;
    const int pbase = blockIdx.x * 256;
    {
        int pp = pbase + t;
        int b = pp >> 13;                       // NS*K = 8192 per batch
        int j = nidx[pp];
        const float* pt = p + ((size_t)b * NN + j) * 3;
        gx[t] = pt[0]; gy[t] = pt[1]; gz[t] = pt[2];
    }
    __syncthreads();
    const float c0 = wce[t * 3 + 0], c1 = wce[t * 3 + 1], c2 = wce[t * 3 + 2];
    const float bv = bias2[t];
    float s1 = 0.f, s2 = 0.f;
    for (int q = 0; q < 256; q++) {
        float h = fmaf(c0, gx[q], fmaf(c1, gy[q], fmaf(c2, gz[q], bv)));
        s1 += h; s2 += h * h;
    }
    atomicAdd(&sum1[t], s1);
    atomicAdd(&ss1[t], s2);
}

// ---------------------------------------------------------------------------
// Build X2 (M x 512, f16): channels [0,256)=pooled max over k, [256,512)=BN1+ReLU(h2)
// Block handles 16 groups (256 positions); thread t == channel t.
// ---------------------------------------------------------------------------
__global__ __launch_bounds__(256) void x2_kernel(const float* __restrict__ p,
                                                 const int* __restrict__ nidx,
                                                 const float* __restrict__ wce,
                                                 const float* __restrict__ bias2,
                                                 const float* __restrict__ sum1,
                                                 const float* __restrict__ ss1,
                                                 const float* __restrict__ g1,
                                                 const float* __restrict__ beta1,
                                                 _Float16* __restrict__ X2) {
    __shared__ float gx[256], gy[256], gz[256];
    const int t = threadIdx.x;
    const int pbase = blockIdx.x * 256;
    {
        int pp = pbase + t;
        int b = pp >> 13;
        int j = nidx[pp];
        const float* pt = p + ((size_t)b * NN + j) * 3;
        gx[t] = pt[0]; gy[t] = pt[1]; gz[t] = pt[2];
    }
    __syncthreads();
    const float c0 = wce[t * 3 + 0], c1 = wce[t * 3 + 1], c2 = wce[t * 3 + 2];
    const float bv = bias2[t];
    const float invM = 1.0f / (float)MTOT;
    const float mean = sum1[t] * invM;
    const float var = ss1[t] * invM - mean * mean;
    const float rs = rsqrtf(var + EPSV);
    const float scale = g1[t] * rs;
    const float shift = beta1[t] - mean * scale;
    for (int grp = 0; grp < 16; grp++) {
        float vals[16];
        float mx = 0.0f;  // ReLU outputs are >= 0
#pragma unroll
        for (int k = 0; k < 16; k++) {
            int q = grp * 16 + k;
            float h = fmaf(c0, gx[q], fmaf(c1, gy[q], fmaf(c2, gz[q], bv)));
            float v = fmaxf(fmaf(h, scale, shift), 0.0f);
            vals[k] = v;
            mx = fmaxf(mx, v);
        }
        size_t pb0 = (size_t)(pbase + grp * 16) * 512;
        _Float16 mh = (_Float16)mx;
#pragma unroll
        for (int k = 0; k < 16; k++) {
            X2[pb0 + (size_t)k * 512 + t]       = mh;               // pooled (broadcast)
            X2[pb0 + (size_t)k * 512 + 256 + t] = (_Float16)vals[k]; // h
        }
    }
}

// ---------------------------------------------------------------------------
// WMMA GEMM: O(Cout x M) = W(Cout x 512) @ X(512 x M), X stored M-major (M x 512 f16).
// Block = 8 waves (256 thr) covering 128 Cout; wave tile = 16 Cout x 64 M, 4 m-iters.
// MODE 0: accumulate per-channel sum/sumsq (BN stats) via shfl + atomics.
// MODE 1: BN+ReLU epilogue, write Y (M x 512 f16).
// MODE 2: BN+ReLU + max-pool over the 16 N-lanes (== one KNN group), write feat.
// ---------------------------------------------------------------------------
template <int MODE>
__global__ __launch_bounds__(256) void gemm512_kernel(const _Float16* __restrict__ Wm,
                                                      const _Float16* __restrict__ Xm,
                                                      float* __restrict__ sumBuf,
                                                      float* __restrict__ ssBuf,
                                                      const float* __restrict__ gam,
                                                      const float* __restrict__ bet,
                                                      _Float16* __restrict__ Y,
                                                      float* __restrict__ feat) {
    const int lane = threadIdx.x & 31;
    const int wv = threadIdx.x >> 5;
    const int lo = lane & 15;
    const int hi = lane >> 4;
    const int co0 = blockIdx.y * 128 + wv * 16;
    const int mblock = blockIdx.x * 256;

    float scale[8], shift[8];
    if constexpr (MODE != 0) {
        const float invM = 1.0f / (float)MTOT;
#pragma unroll
        for (int r = 0; r < 8; r++) {
            int c = co0 + hi * 8 + r;
            float mean = sumBuf[c] * invM;
            float var = ssBuf[c] * invM - mean * mean;
            float rs = rsqrtf(var + EPSV);
            scale[r] = gam[c] * rs;
            shift[r] = bet[c] - mean * scale[r];
        }
    }
    float s1[8] = {0, 0, 0, 0, 0, 0, 0, 0};
    float s2[8] = {0, 0, 0, 0, 0, 0, 0, 0};

    // A fragment row: lanes 0-15 and 16-31 both hold rows M=0..15; hi selects K halves.
    const _Float16* wrow = Wm + (size_t)(co0 + lo) * 512 + hi * 16;

    for (int mi = 0; mi < 4; mi++) {
        const int mbase = mblock + mi * 64;
        const f8 zero = {0.f, 0.f, 0.f, 0.f, 0.f, 0.f, 0.f, 0.f};
        f8 acc[4];
        const _Float16* xr[4];
#pragma unroll
        for (int ti = 0; ti < 4; ti++) {
            acc[ti] = zero;
            xr[ti] = Xm + (size_t)(mbase + ti * 16 + lo) * 512 + hi * 16;
        }
#pragma unroll 4
        for (int kc = 0; kc < 16; kc++) {
            const int k0 = kc * 32;
            h8 al = *(const h8*)(wrow + k0);
            h8 ah = *(const h8*)(wrow + k0 + 8);
            h16 a = __builtin_shufflevector(al, ah, 0, 1, 2, 3, 4, 5, 6, 7, 8, 9, 10, 11, 12, 13, 14, 15);
#pragma unroll
            for (int ti = 0; ti < 4; ti++) {
                h8 bl = *(const h8*)(xr[ti] + k0);
                h8 bh = *(const h8*)(xr[ti] + k0 + 8);
                h16 bm = __builtin_shufflevector(bl, bh, 0, 1, 2, 3, 4, 5, 6, 7, 8, 9, 10, 11, 12, 13, 14, 15);
                acc[ti] = __builtin_amdgcn_wmma_f32_16x16x32_f16(false, a, false, bm,
                                                                 (short)0, acc[ti], false, false);
            }
        }

        if constexpr (MODE == 0) {
#pragma unroll
            for (int ti = 0; ti < 4; ti++)
#pragma unroll
                for (int r = 0; r < 8; r++) {
                    float v = acc[ti][r];
                    s1[r] += v;
                    s2[r] += v * v;
                }
        } else if constexpr (MODE == 1) {
#pragma unroll
            for (int ti = 0; ti < 4; ti++) {
                int m = mbase + ti * 16 + lo;
                h8 oh;
#pragma unroll
                for (int r = 0; r < 8; r++) {
                    float v = fmaxf(fmaf(acc[ti][r], scale[r], shift[r]), 0.0f);
                    oh[r] = (_Float16)v;
                }
                *(h8*)(Y + (size_t)m * 512 + co0 + hi * 8) = oh;
            }
        } else {
#pragma unroll
            for (int ti = 0; ti < 4; ti++) {
                float v[8];
#pragma unroll
                for (int r = 0; r < 8; r++)
                    v[r] = fmaxf(fmaf(acc[ti][r], scale[r], shift[r]), 0.0f);
#pragma unroll
                for (int r = 0; r < 8; r++) {
                    v[r] = fmaxf(v[r], __shfl_xor(v[r], 1, 32));
                    v[r] = fmaxf(v[r], __shfl_xor(v[r], 2, 32));
                    v[r] = fmaxf(v[r], __shfl_xor(v[r], 4, 32));
                    v[r] = fmaxf(v[r], __shfl_xor(v[r], 8, 32));
                }
                if (lo == 0) {
                    int g = (mbase >> 4) + ti;   // group index == b*NS + s
                    int b = g >> 9;              // NS = 512
                    int sI = g & 511;
#pragma unroll
                    for (int r = 0; r < 8; r++) {
                        int c = co0 + hi * 8 + r;
                        feat[(size_t)b * 256 * NSAMP + (size_t)c * NSAMP + sI] = v[r];
                    }
                }
            }
        }
    }

    if constexpr (MODE == 0) {
#pragma unroll
        for (int r = 0; r < 8; r++) {
            float a1 = s1[r], a2 = s2[r];
            a1 += __shfl_xor(a1, 1, 32); a2 += __shfl_xor(a2, 1, 32);
            a1 += __shfl_xor(a1, 2, 32); a2 += __shfl_xor(a2, 2, 32);
            a1 += __shfl_xor(a1, 4, 32); a2 += __shfl_xor(a2, 4, 32);
            a1 += __shfl_xor(a1, 8, 32); a2 += __shfl_xor(a2, 8, 32);
            if (lo == 0) {
                atomicAdd(&sumBuf[co0 + hi * 8 + r], a1);
                atomicAdd(&ssBuf[co0 + hi * 8 + r], a2);
            }
        }
    }
}

// ---------------------------------------------------------------------------
extern "C" void kernel_launch(void* const* d_in, const int* in_sizes, int n_in,
                              void* d_out, int out_size, void* d_ws, size_t ws_size,
                              hipStream_t stream) {
    const float* p    = (const float*)d_in[0];
    const float* w1   = (const float*)d_in[1];
    const float* w2   = (const float*)d_in[2];
    const float* b2   = (const float*)d_in[3];
    const float* g1   = (const float*)d_in[4];
    const float* bb1  = (const float*)d_in[5];
    const float* w3   = (const float*)d_in[6];
    const float* g3   = (const float*)d_in[7];
    const float* bb3  = (const float*)d_in[8];
    const float* w4   = (const float*)d_in[9];
    const float* g4   = (const float*)d_in[10];
    const float* bb4  = (const float*)d_in[11];

    float* out = (float*)d_out;
    float* cntrd = out;                              // (B, NS, 3)
    float* feat = out + (size_t)BB * NSAMP * 3;      // (B, 256, NS)

    char* ws = (char*)d_ws;
    size_t off = 0;
    _Float16* X2 = (_Float16*)(ws + off);  off += (size_t)MTOT * 512 * sizeof(_Float16);
    _Float16* X3 = (_Float16*)(ws + off);  off += (size_t)MTOT * 512 * sizeof(_Float16);
    _Float16* w3h = (_Float16*)(ws + off); off += (size_t)512 * 512 * sizeof(_Float16);
    _Float16* w4h = (_Float16*)(ws + off); off += (size_t)256 * 512 * sizeof(_Float16);
    int* nidx = (int*)(ws + off);          off += (size_t)MTOT * sizeof(int);
    float* wce = (float*)(ws + off);       off += 4096;
    float* stats = (float*)(ws + off);     off += 2048 * sizeof(float);
    float* sum1 = stats;         float* ss1 = stats + 256;
    float* sum3 = stats + 512;   float* ss3 = stats + 1024;
    float* sum4 = stats + 1536;  float* ss4 = stats + 1792;

    zero_kernel<<<8, 256, 0, stream>>>(stats, 2048);
    wce_kernel<<<1, 256, 0, stream>>>(w1, w2, wce);
    cvt_kernel<<<1024, 256, 0, stream>>>(w3, w3h, 512 * 512);
    cvt_kernel<<<512, 256, 0, stream>>>(w4, w4h, 256 * 512);

    fps_kernel<<<32, 256, 0, stream>>>(p, cntrd);
    knn_kernel<<<1024, 256, 0, stream>>>(p, cntrd, nidx);

    stats1_kernel<<<MTOT / 256, 256, 0, stream>>>(p, nidx, wce, b2, sum1, ss1);
    x2_kernel<<<GTOT / 16, 256, 0, stream>>>(p, nidx, wce, b2, sum1, ss1, g1, bb1, X2);

    // conv block 2a: 512 -> 512 (w3), BN3+ReLU -> X3
    dim3 grid3(MTOT / 256, 4);
    gemm512_kernel<0><<<grid3, 256, 0, stream>>>(w3h, X2, sum3, ss3, nullptr, nullptr, nullptr, nullptr);
    gemm512_kernel<1><<<grid3, 256, 0, stream>>>(w3h, X2, sum3, ss3, g3, bb3, X3, nullptr);

    // conv block 2b: 512 -> 256 (w4), BN4+ReLU, max over k -> feat
    dim3 grid4(MTOT / 256, 2);
    gemm512_kernel<0><<<grid4, 256, 0, stream>>>(w4h, X3, sum4, ss4, nullptr, nullptr, nullptr, nullptr);
    gemm512_kernel<2><<<grid4, 256, 0, stream>>>(w4h, X3, sum4, ss4, g4, bb4, nullptr, feat);
}
